// CategoryHeteroGNN_68401649156707
// MI455X (gfx1250) — matrix operand order repaired and checked by reading, not compiled, not run on hardware
//
#include <hip/hip_runtime.h>
#include <hip/hip_bf16.h>

typedef __attribute__((ext_vector_type(16))) _Float16 v16h;
typedef __attribute__((ext_vector_type(8)))  _Float16 v8h;
typedef __attribute__((ext_vector_type(8)))  float    v8f;

#define HID 64

// ---------------- elementwise / prep kernels ----------------

__global__ void k_fill(float* p, float v, int n) {
    int i = blockIdx.x * blockDim.x + threadIdx.x;
    if (i < n) p[i] = v;
}

__global__ void k_deg(const int* __restrict__ dst, float* __restrict__ deg, int E) {
    int e = blockIdx.x * blockDim.x + threadIdx.x;
    if (e < E) atomicAdd(&deg[dst[e]], 1.0f);
}

__global__ void k_rsqrt(float* p, int n) {
    int i = blockIdx.x * blockDim.x + threadIdx.x;
    if (i < n) p[i] = rsqrtf(p[i]);   // deg >= 1 always (self loop)
}

__global__ void k_cvt(const float* __restrict__ in, _Float16* __restrict__ out, int n) {
    int i = blockIdx.x * blockDim.x + threadIdx.x;
    if (i < n) out[i] = (_Float16)in[i];
}

// 64x64 weight: convert f32 -> f16 AND transpose (out[n,k] = in[k,n]) so the
// WMMA B-fragment loads become contiguous 32B vector loads.
__global__ void k_cvt_wT(const float* __restrict__ in, _Float16* __restrict__ out) {
    int i = blockIdx.x * blockDim.x + threadIdx.x;   // i < 4096
    int k = i >> 6, n = i & 63;
    out[n * HID + k] = (_Float16)in[k * HID + n];
}

__global__ void k_relu_cvt(const float* __restrict__ in, _Float16* __restrict__ out, int n) {
    int i = blockIdx.x * blockDim.x + threadIdx.x;
    if (i < n) out[i] = (_Float16)fmaxf(in[i], 0.0f);
}

// ---------------- dual WMMA GEMM: Ha = X@Wa, Hb = X@Wb (Wa/Wb given transposed)
// Block = 256 threads = 8 waves; one 16-row strip of X per block (staged in LDS).
// Wave w: colTile = w&3, matrix = w>>2. Each wave: 2 x v_wmma_f32_16x16x32_f16.

__global__ void __launch_bounds__(256)
k_gemm_dual(const _Float16* __restrict__ X,
            const _Float16* __restrict__ WaT, const _Float16* __restrict__ WbT,
            float* __restrict__ Ha, float* __restrict__ Hb)
{
    __shared__ __align__(16) _Float16 As[16 * HID];   // 2 KB: 16 rows x 64 halves
    const int t = threadIdx.x;
    const size_t row0 = (size_t)blockIdx.x * 16;

    // cooperative stage of the A strip: 2048 bytes, 8 bytes per thread
    ((uint2*)As)[t] = ((const uint2*)(X + row0 * HID))[t];
    __syncthreads();

    const int lane = t & 31;
    const int wave = t >> 5;
    const int colT = wave & 3;
    const int mat  = wave >> 2;
    const _Float16* __restrict__ WT = mat ? WbT : WaT;
    float* __restrict__ H           = mat ? Hb : Ha;

    const int m   = lane & 15;       // A row within tile / B-C column within tile
    const int hi  = lane >> 4;
    const int kbA = hi * 16;         // A: lanes16-31 hold K=16..31
    const int kbB = hi * 16;         // B: lanes16-31 hold K=16..31
    const int col = colT * 16 + m;   // output column this lane owns in B/C

    v8f c = {};
#pragma unroll
    for (int kc = 0; kc < HID; kc += 32) {
        // A fragment: two contiguous 16B runs from LDS
        v8h a_lo = *(const v8h*)(As + m * HID + kc + kbA);
        v8h a_hi = *(const v8h*)(As + m * HID + kc + 8 + kbA);
        v16h a = __builtin_shufflevector(a_lo, a_hi,
                                         0,1,2,3,4,5,6,7,8,9,10,11,12,13,14,15);
        // B fragment: 16 K-consecutive halves of column `col` = one contiguous
        // 32B run of the transposed weight matrix
        v16h b = *(const v16h*)(WT + (size_t)col * HID + kc + kbB);

        c = __builtin_amdgcn_wmma_f32_16x16x32_f16(
                /*neg_a=*/false, a, /*neg_b=*/false, b,
                /*c_mod=*/(short)0, c, /*reuse_a=*/false, /*reuse_b=*/false);
    }

    // C layout: VGPR j -> row (hi*8 + j), col m
#pragma unroll
    for (int j = 0; j < 8; ++j)
        H[(row0 + hi * 8 + j) * HID + col] = c[j];
}

// ---------------- aggregation ----------------

// acc[i,f] = bs[f] + bd[f] + dinv_s[i]^2 * Hs[i,f] + dinv_d[i]^2 * Hd[i,f]
__global__ void k_init_acc(float* __restrict__ acc,
                           const float* __restrict__ Hs, const float* __restrict__ Hd,
                           const float* __restrict__ dvs, const float* __restrict__ dvd,
                           const float* __restrict__ bs, const float* __restrict__ bd,
                           int n64)
{
    int t = blockIdx.x * blockDim.x + threadIdx.x;
    if (t >= n64) return;
    int i = t >> 6, f = t & 63;
    float a = dvs[i], b = dvd[i];
    acc[t] = bs[f] + bd[f] + a * a * Hs[t] + b * b * Hd[t];
}

// 64 lanes per edge: coalesced gather of H[src] + coalesced atomic scatter to acc[dst]
__global__ void k_scatter(const int* __restrict__ ei, int E,
                          const float* __restrict__ dinv,
                          const float* __restrict__ H, float* __restrict__ acc)
{
    long long t = (long long)blockIdx.x * blockDim.x + threadIdx.x;
    int e = (int)(t >> 6);
    if (e >= E) return;
    int f = (int)(t & 63);
    int s = ei[e];
    int d = ei[E + e];
    float w = dinv[s] * dinv[d];
    atomicAdd(&acc[(size_t)d * HID + f], w * H[(size_t)s * HID + f]);
}

// out = relu(acc) @ Wlin + blin  (64 -> 3)
__global__ void k_final(const float* __restrict__ acc,
                        const float* __restrict__ Wl, const float* __restrict__ bl,
                        float* __restrict__ out, int Nn)
{
    int i = blockIdx.x * blockDim.x + threadIdx.x;
    if (i >= Nn) return;
    float a0 = bl[0], a1 = bl[1], a2 = bl[2];
    const float* row = acc + (size_t)i * HID;
#pragma unroll
    for (int f = 0; f < HID; ++f) {
        float v = fmaxf(row[f], 0.0f);
        a0 += v * Wl[f * 3 + 0];
        a1 += v * Wl[f * 3 + 1];
        a2 += v * Wl[f * 3 + 2];
    }
    out[i * 3 + 0] = a0;
    out[i * 3 + 1] = a1;
    out[i * 3 + 2] = a2;
}

// ---------------- launch ----------------

extern "C" void kernel_launch(void* const* d_in, const int* in_sizes, int n_in,
                              void* d_out, int out_size, void* d_ws, size_t ws_size,
                              hipStream_t stream)
{
    (void)n_in; (void)out_size; (void)ws_size;
    const float* x    = (const float*)d_in[0];
    const int*   ei_s = (const int*)  d_in[1];
    const int*   ei_d = (const int*)  d_in[2];
    const float* W1s  = (const float*)d_in[3];
    const float* b1s  = (const float*)d_in[4];
    const float* W1d  = (const float*)d_in[5];
    const float* b1d  = (const float*)d_in[6];
    const float* W2s  = (const float*)d_in[7];
    const float* b2s  = (const float*)d_in[8];
    const float* W2d  = (const float*)d_in[9];
    const float* b2d  = (const float*)d_in[10];
    const float* Wlin = (const float*)d_in[11];
    const float* blin = (const float*)d_in[12];
    float* out = (float*)d_out;

    const int Nn = in_sizes[0] / HID;   // 50000 (multiple of 16)
    const int E  = in_sizes[1] / 2;     // 800000

    // carve scratch (~45 MB)
    char* ws = (char*)d_ws;
    size_t off = 0;
    auto carve = [&](size_t bytes) -> void* {
        void* p = ws + off;
        off = (off + bytes + 255) & ~(size_t)255;
        return p;
    };
    float*    dinv_s = (float*)   carve((size_t)Nn * 4);
    float*    dinv_d = (float*)   carve((size_t)Nn * 4);
    _Float16* xh     = (_Float16*)carve((size_t)Nn * HID * 2);  // reused as h1 (f16)
    _Float16* wh1s   = (_Float16*)carve(HID * HID * 2);         // transposed f16 weights
    _Float16* wh1d   = (_Float16*)carve(HID * HID * 2);
    _Float16* wh2s   = (_Float16*)carve(HID * HID * 2);
    _Float16* wh2d   = (_Float16*)carve(HID * HID * 2);
    float*    Hs     = (float*)   carve((size_t)Nn * HID * 4);
    float*    Hd     = (float*)   carve((size_t)Nn * HID * 4);
    float*    acc    = (float*)   carve((size_t)Nn * HID * 4);

    const int B = 256;
    auto cdiv = [](long long a, long long b) { return (int)((a + b - 1) / b); };
    const int n64 = Nn * HID;

    // degrees (init 1.0 for self loops), then rsqrt in place
    k_fill<<<cdiv(Nn, B), B, 0, stream>>>(dinv_s, 1.0f, Nn);
    k_fill<<<cdiv(Nn, B), B, 0, stream>>>(dinv_d, 1.0f, Nn);
    k_deg<<<cdiv(E, B), B, 0, stream>>>(ei_s + E, dinv_s, E);
    k_deg<<<cdiv(E, B), B, 0, stream>>>(ei_d + E, dinv_d, E);
    k_rsqrt<<<cdiv(Nn, B), B, 0, stream>>>(dinv_s, Nn);
    k_rsqrt<<<cdiv(Nn, B), B, 0, stream>>>(dinv_d, Nn);

    // f16 conversions (X plain, weights transposed)
    k_cvt<<<cdiv(n64, B), B, 0, stream>>>(x, xh, n64);
    k_cvt_wT<<<cdiv(HID * HID, B), B, 0, stream>>>(W1s, wh1s);
    k_cvt_wT<<<cdiv(HID * HID, B), B, 0, stream>>>(W1d, wh1d);
    k_cvt_wT<<<cdiv(HID * HID, B), B, 0, stream>>>(W2s, wh2s);
    k_cvt_wT<<<cdiv(HID * HID, B), B, 0, stream>>>(W2d, wh2d);

    // ---- layer 1 ----
    k_gemm_dual<<<Nn / 16, 256, 0, stream>>>(xh, wh1s, wh1d, Hs, Hd);
    k_init_acc<<<cdiv(n64, B), B, 0, stream>>>(acc, Hs, Hd, dinv_s, dinv_d, b1s, b1d, n64);
    k_scatter<<<cdiv((long long)E * HID, B), B, 0, stream>>>(ei_s, E, dinv_s, Hs, acc);
    k_scatter<<<cdiv((long long)E * HID, B), B, 0, stream>>>(ei_d, E, dinv_d, Hd, acc);
    k_relu_cvt<<<cdiv(n64, B), B, 0, stream>>>(acc, xh, n64);   // h1 -> f16, reuse xh

    // ---- layer 2 ----
    k_gemm_dual<<<Nn / 16, 256, 0, stream>>>(xh, wh2s, wh2d, Hs, Hd);
    k_init_acc<<<cdiv(n64, B), B, 0, stream>>>(acc, Hs, Hd, dinv_s, dinv_d, b2s, b2d, n64);
    k_scatter<<<cdiv((long long)E * HID, B), B, 0, stream>>>(ei_s, E, dinv_s, Hs, acc);
    k_scatter<<<cdiv((long long)E * HID, B), B, 0, stream>>>(ei_d, E, dinv_d, Hd, acc);

    // ---- head ----
    k_final<<<cdiv(Nn, B), B, 0, stream>>>(acc, Wlin, blin, out, Nn);
}
